// BruteForce_85048942395817
// MI455X (gfx1250) — compile-verified, matrix operand order checked by reading, not run on hardware
//
#include <hip/hip_runtime.h>

// Problem sizes (fixed by the reference)
#define BQ 256        // queries (M)
#define NC 500000     // candidates (N), = 16 * 31250 exactly
#define DD 128        // dims (K)
#define KK 100        // top-k

#define CT_PER_BLOCK 64   // 16-wide candidate tiles per block (1024 candidates)
#define GEMM_THREADS 512  // 16 waves; wave w owns query tile w
#define TILE_BYTES   8192 // per-tile bf16 block: 4KB hi plane + 4KB lo plane
#define TILE_ELEMS   4096 // in __bf16 units

typedef __attribute__((ext_vector_type(16))) __bf16 v16bf;
typedef __attribute__((ext_vector_type(8)))  __bf16 v8bf;
typedef __attribute__((ext_vector_type(8)))  float  v8f;
typedef __attribute__((ext_vector_type(4)))  float  v4f;
typedef __attribute__((ext_vector_type(4)))  int    v4i;

union Frag16 { v16bf v; v8bf h[2]; };

// ---- CDNA5 async global->LDS staging (guarded; sync fallback keeps compile safe)
#if defined(__has_builtin)
# if __has_builtin(__builtin_amdgcn_global_load_async_to_lds_b128)
#  define HAVE_ASYNC_LDS 1
# endif
#endif
#ifndef HAVE_ASYNC_LDS
# define HAVE_ASYNC_LDS 0
#endif

#define AS1 __attribute__((address_space(1)))
#define AS3 __attribute__((address_space(3)))

__device__ __forceinline__ void stage16(const void* g, void* l) {
#if HAVE_ASYNC_LDS
  // Builtin signature (from hipcc diagnostic): (int4* global, int4* lds, Ii, Ii)
  __builtin_amdgcn_global_load_async_to_lds_b128((AS1 v4i*)g, (AS3 v4i*)l, 0, 0);
#else
  *(v8bf*)l = *(const v8bf*)g;   // synchronous fallback
#endif
}

__device__ __forceinline__ void wait_async0() {
#if HAVE_ASYNC_LDS
# if __has_builtin(__builtin_amdgcn_s_wait_asynccnt)
  __builtin_amdgcn_s_wait_asynccnt(0);
# else
  asm volatile("s_wait_asynccnt 0x0" ::: "memory");
# endif
#endif
}

// ---------------------------------------------------------------------------
// Pass 0a: split fp32 queries into bf16 hi/lo planes (x ~= hi + lo)
// ---------------------------------------------------------------------------
__global__ __launch_bounds__(256)
void convert_queries_kernel(const float* __restrict__ q,
                            __bf16* __restrict__ qhi,
                            __bf16* __restrict__ qlo) {
  int i = blockIdx.x * blockDim.x + threadIdx.x;
  if (i < BQ * DD) {
    float f = q[i];
    __bf16 h = (__bf16)f;
    qhi[i] = h;
    qlo[i] = (__bf16)(f - (float)h);
  }
}

// ---------------------------------------------------------------------------
// Pass 0b: split fp32 candidates into tile-blocked bf16 hi/lo planes.
// Tile i (16 candidates) -> 8KB block: [hi 16x128 bf16][lo 16x128 bf16].
// Pure bandwidth pass (~512MB total) so the GEMM stage is a raw byte copy.
// ---------------------------------------------------------------------------
__global__ __launch_bounds__(256)
void convert_candidates_kernel(const float* __restrict__ cand,
                               __bf16* __restrict__ cbf) {
  const size_t e = ((size_t)blockIdx.x * 256 + threadIdx.x) * 4;  // 4 f32/thread
  if (e >= (size_t)NC * DD) return;
  const v4f c4 = *(const v4f*)(cand + e);
  const size_t n  = e >> 7;          // candidate row (d = e & 127 contiguous)
  const int    d  = (int)(e & 127);
  const size_t tl = n >> 4;
  const int    nl = (int)(n & 15);
  __bf16* hi = cbf + tl * TILE_ELEMS + (size_t)nl * DD + d;
  __bf16* lo = hi + 2048;            // lo plane 4KB after hi plane
#pragma unroll
  for (int j = 0; j < 4; ++j) {
    const float f = c4[j];
    const __bf16 h = (__bf16)f;
    hi[j] = h;
    lo[j] = (__bf16)(f - (float)h);
  }
}

// ---------------------------------------------------------------------------
// Pass 1: scores = Q * C^T via bf16x2 WMMA, async double-buffered LDS staging.
//   D = Ahi*Bhi + Ahi*Blo + Alo*Bhi   (3x v_wmma_f32_16x16x32_bf16 per K=32)
// Each block: all 256 queries x 1024 candidates; candidates stream once.
// ---------------------------------------------------------------------------
__global__ __launch_bounds__(GEMM_THREADS)
void gemm_bf16x2_kernel(const __bf16* __restrict__ cbf,
                        const __bf16* __restrict__ qhi,
                        const __bf16* __restrict__ qlo,
                        float* __restrict__ scores) {
  __shared__ __align__(16) char buf[2][TILE_BYTES];   // 16 KB double buffer

  const int tid  = threadIdx.x;
  const int lane = tid & 31;
  const int wave = tid >> 5;          // 0..15 => query tile
  const int nlo  = lane & 15;         // column within 16 (N for B, M for A)
  const int hsel = lane >> 4;         // half-select per 16-bit fragment layout
  const int mrow = (wave << 4) + nlo; // query row this lane's A-fragment holds

  // Resident A fragments: 4 K-chunks of 32, hi + lo planes.
  Frag16 aHi[4], aLo[4];
#pragma unroll
  for (int kc = 0; kc < 4; ++kc) {
    const int b0 = kc * 32 + hsel * 8;
    const __bf16* ph = qhi + (size_t)mrow * DD + b0;
    const __bf16* pl = qlo + (size_t)mrow * DD + b0;
    aHi[kc].h[0] = *(const v8bf*)(ph);
    aHi[kc].h[1] = *(const v8bf*)(ph + 16);
    aLo[kc].h[0] = *(const v8bf*)(pl);
    aLo[kc].h[1] = *(const v8bf*)(pl + 16);
  }

  const int NT    = NC / 16;                 // 31250 candidate tiles total
  const int tile0 = blockIdx.x * CT_PER_BLOCK;
  const int tend  = (NT - tile0 < CT_PER_BLOCK) ? (NT - tile0) : CT_PER_BLOCK;

  const size_t myOff = (size_t)tid * 16;     // 512 threads x 16B = one 8KB tile

  // Prime the pipeline: tile 0 -> buf[0]
  stage16((const char*)(cbf + (size_t)tile0 * TILE_ELEMS) + myOff, buf[0] + myOff);

  for (int t = 0; t < tend; ++t) {
    const int tile = tile0 + t;
    const int n0   = tile << 4;
    const int cur  = t & 1;

    wait_async0();       // this wave's slice of tile t has landed in LDS
    __syncthreads();     // all waves' slices landed; prev compute done (buffer reuse safe)

    if (t + 1 < tend)    // overlap: fetch tile t+1 while computing on tile t
      stage16((const char*)(cbf + (size_t)(tile + 1) * TILE_ELEMS) + myOff,
              buf[cur ^ 1] + myOff);

    const __bf16* sHi = (const __bf16*)(buf[cur]);
    const __bf16* sLo = (const __bf16*)(buf[cur] + 4096);

    v8f acc = {};
#pragma unroll
    for (int kc = 0; kc < 4; ++kc) {
      Frag16 bHi, bLo;
      const int b0 = kc * 32 + hsel * 8;
      const __bf16* ph = &sHi[nlo * DD + b0];
      const __bf16* pl = &sLo[nlo * DD + b0];
      bHi.h[0] = *(const v8bf*)(ph);
      bHi.h[1] = *(const v8bf*)(ph + 16);
      bLo.h[0] = *(const v8bf*)(pl);
      bLo.h[1] = *(const v8bf*)(pl + 16);

      acc = __builtin_amdgcn_wmma_f32_16x16x32_bf16(
          false, aHi[kc].v, false, bHi.v, (short)0, acc, false, false);
      acc = __builtin_amdgcn_wmma_f32_16x16x32_bf16(
          false, aHi[kc].v, false, bLo.v, (short)0, acc, false, false);
      acc = __builtin_amdgcn_wmma_f32_16x16x32_bf16(
          false, aLo[kc].v, false, bHi.v, (short)0, acc, false, false);
    }

    // C/D layout: VGPR r, lanes 0-15 -> M = r, lanes 16-31 -> M = r+8; N = lane%16.
    const int ncol = n0 + nlo;
#pragma unroll
    for (int r = 0; r < 8; ++r) {
      const int m = (wave << 4) + r + (hsel << 3);
      scores[(size_t)m * NC + ncol] = acc[r];
    }
  }
}

// ---------------------------------------------------------------------------
// Pass 2: per-query top-100.
// One block per query. Each thread keeps a sorted top-16 of its strided slice,
// then 100 rounds of LDS extract-max over the 256*16 = 4096 survivors.
// ---------------------------------------------------------------------------
#define TL 16

__global__ __launch_bounds__(256)
void topk_kernel(const float* __restrict__ scores,
                 const long long* __restrict__ ident,
                 float* __restrict__ outVals,
                 long long* __restrict__ outIds) {
  __shared__ float sv[256 * TL];
  __shared__ int   si[256 * TL];
  __shared__ float rv[256];
  __shared__ int   rs[256];

  const int b = blockIdx.x;
  const int t = threadIdx.x;
  const float* __restrict__ row = scores + (size_t)b * NC;

  float lv[TL];
  int   li[TL];
#pragma unroll
  for (int j = 0; j < TL; ++j) { lv[j] = -__builtin_inff(); li[j] = 0; }

  for (int n = t; n < NC; n += 256) {
    const float s = row[n];
    if (s > lv[TL - 1]) {               // rare: insert + single bubble pass
      lv[TL - 1] = s; li[TL - 1] = n;
#pragma unroll
      for (int j = TL - 1; j > 0; --j) {
        if (lv[j] > lv[j - 1]) {
          float tv = lv[j]; lv[j] = lv[j - 1]; lv[j - 1] = tv;
          int   ti = li[j]; li[j] = li[j - 1]; li[j - 1] = ti;
        }
      }
    }
  }
#pragma unroll
  for (int j = 0; j < TL; ++j) { sv[t * TL + j] = lv[j]; si[t * TL + j] = li[j]; }
  __syncthreads();

  for (int r = 0; r < KK; ++r) {
    float best = -__builtin_inff();
    int bslot = 0;
#pragma unroll
    for (int j = 0; j < TL; ++j) {
      const int slot = j * 256 + t;
      const float v = sv[slot];
      if (v > best) { best = v; bslot = slot; }
    }
    rv[t] = best; rs[t] = bslot;
    __syncthreads();
    for (int off = 128; off > 0; off >>= 1) {
      if (t < off && rv[t + off] > rv[t]) { rv[t] = rv[t + off]; rs[t] = rs[t + off]; }
      __syncthreads();
    }
    if (t == 0) {
      const int w = rs[0];
      outVals[(size_t)b * KK + r] = rv[0];
      outIds [(size_t)b * KK + r] = ident[si[w]];
      sv[w] = -__builtin_inff();
    }
    __syncthreads();
  }
}

// ---------------------------------------------------------------------------
extern "C" void kernel_launch(void* const* d_in, const int* in_sizes, int n_in,
                              void* d_out, int out_size, void* d_ws, size_t ws_size,
                              hipStream_t stream) {
  const float*     queries = (const float*)d_in[0];      // [256][128] f32
  const float*     cand    = (const float*)d_in[1];      // [500000][128] f32
  const long long* ident   = (const long long*)d_in[2];  // [500000] i64
  (void)in_sizes; (void)n_in; (void)out_size; (void)ws_size;

  // Workspace layout: scores (512 MB) | qhi (64 KB) | qlo (64 KB) | cbf (256 MB)
  char* ws = (char*)d_ws;
  float* scores = (float*)ws;
  const size_t scoresBytes = (size_t)BQ * (size_t)NC * sizeof(float);
  __bf16* qhi = (__bf16*)(ws + scoresBytes);
  __bf16* qlo = qhi + (size_t)BQ * DD;
  __bf16* cbf = qlo + (size_t)BQ * DD;   // tile-blocked candidate planes

  // Output: values f32 [256][100], then ids i64 [256][100]
  float*     outVals = (float*)d_out;
  long long* outIds  = (long long*)((char*)d_out + (size_t)BQ * KK * sizeof(float));

  convert_queries_kernel<<<(BQ * DD + 255) / 256, 256, 0, stream>>>(queries, qhi, qlo);

  const size_t totalElems = (size_t)NC * DD;                     // 64M
  const int convBlocks = (int)((totalElems / 4 + 255) / 256);    // 4 f32 per thread
  convert_candidates_kernel<<<convBlocks, 256, 0, stream>>>(cand, cbf);

  const int NT      = NC / 16;                                   // 31250
  const int nblocks = (NT + CT_PER_BLOCK - 1) / CT_PER_BLOCK;    // 489
  gemm_bf16x2_kernel<<<nblocks, GEMM_THREADS, 0, stream>>>(cbf, qhi, qlo, scores);

  topk_kernel<<<BQ, 256, 0, stream>>>(scores, ident, outVals, outIds);
}
